// GATBlock_57621281243370
// MI455X (gfx1250) — matrix operand order repaired and checked by reading, not compiled, not run on hardware
//
#include <hip/hip_runtime.h>
#include <cstdint>
#include <cstddef>

// ---------------------------------------------------------------------------
// GAT block for MI455X (gfx1250, wave32).
//   h = x @ W^T          -> v_wmma_f32_16x16x32_f16 (f32 accumulate)
//   a_s/a_d              -> wave-per-node dot products, shuffle reduce
//   softmax over edges   -> unshifted (logits provably bounded), exp clamped
//   scatter aggregation  -> global_atomic_add_f32 into d_out (L2-resident)
//   epilogue             -> ELU + bias + residual, in place on d_out
// ---------------------------------------------------------------------------

typedef __attribute__((ext_vector_type(16))) _Float16 v16h;
typedef __attribute__((ext_vector_type(8)))  float    v8f;

#define F_IN   256
#define HC     256           // HEADS * C_OUT
#define HEADS  4
#define NEG_SLOPE 0.2f

// ---------------------------------------------------------------- init ------
__global__ void gat_init(float* __restrict__ out, float* __restrict__ denom,
                         int total, int nd) {
  int i = blockIdx.x * blockDim.x + threadIdx.x;
  if (i < total) out[i] = 0.0f;
  if (i < nd)    denom[i] = 0.0f;
}

// ------------------------------------------------------- WMMA GEMM ---------
// One wave computes one 16x16 tile of h = x @ W^T.
// Tiles: mtiles = ceil(N/16) along rows, 16 tiles along the 256 output cols.
__global__ void gat_gemm_wmma(const float* __restrict__ x,
                              const float* __restrict__ W,
                              float* __restrict__ h, int N) {
  const int gid    = blockIdx.x * blockDim.x + threadIdx.x;
  const int wave   = gid >> 5;
  const int lane   = threadIdx.x & 31;
  const int mtiles = (N + 15) >> 4;
  const int mt = wave >> 4;            // row tile
  const int nt = wave & 15;            // col tile (HC/16 == 16)
  if (mt >= mtiles) return;

  const int m0  = mt << 4;
  const int n0  = nt << 4;
  const int grp = lane >> 4;           // 0: lanes 0-15, 1: lanes 16-31
  const int lm  = lane & 15;

  int arow = m0 + lm; if (arow >= N) arow = N - 1;   // clamp; stores guarded
  const float* __restrict__ aptr = x + (size_t)arow * F_IN;
  const float* __restrict__ bptr = W + (size_t)(n0 + lm) * F_IN; // B = W^T col

  v8f c = {0.f, 0.f, 0.f, 0.f, 0.f, 0.f, 0.f, 0.f};

#pragma unroll
  for (int k0 = 0; k0 < F_IN; k0 += 32) {
    v16h a, b;
    // A fragment (16x32 f16): lanes 0-15 hold K=[0..7]&[16..23],
    //                         lanes 16-31 hold K=[8..15]&[24..31]
    const float4 a0 = *(const float4*)(aptr + k0 + grp * 8);
    const float4 a1 = *(const float4*)(aptr + k0 + grp * 8 + 4);
    const float4 a2 = *(const float4*)(aptr + k0 + 16 + grp * 8);
    const float4 a3 = *(const float4*)(aptr + k0 + 16 + grp * 8 + 4);
    a[0]=(_Float16)a0.x;  a[1]=(_Float16)a0.y;  a[2]=(_Float16)a0.z;  a[3]=(_Float16)a0.w;
    a[4]=(_Float16)a1.x;  a[5]=(_Float16)a1.y;  a[6]=(_Float16)a1.z;  a[7]=(_Float16)a1.w;
    a[8]=(_Float16)a2.x;  a[9]=(_Float16)a2.y;  a[10]=(_Float16)a2.z; a[11]=(_Float16)a2.w;
    a[12]=(_Float16)a3.x; a[13]=(_Float16)a3.y; a[14]=(_Float16)a3.z; a[15]=(_Float16)a3.w;
    // B fragment (32x16 f16): lanes 0-15 hold K=[0..7]&[16..23],
    //                         lanes 16-31 hold K=[8..15]&[24..31]
    const float4 b0 = *(const float4*)(bptr + k0 + grp * 8);
    const float4 b1 = *(const float4*)(bptr + k0 + grp * 8 + 4);
    const float4 b2 = *(const float4*)(bptr + k0 + 16 + grp * 8);
    const float4 b3 = *(const float4*)(bptr + k0 + 16 + grp * 8 + 4);
    b[0]=(_Float16)b0.x;  b[1]=(_Float16)b0.y;  b[2]=(_Float16)b0.z;  b[3]=(_Float16)b0.w;
    b[4]=(_Float16)b1.x;  b[5]=(_Float16)b1.y;  b[6]=(_Float16)b1.z;  b[7]=(_Float16)b1.w;
    b[8]=(_Float16)b2.x;  b[9]=(_Float16)b2.y;  b[10]=(_Float16)b2.z; b[11]=(_Float16)b2.w;
    b[12]=(_Float16)b3.x; b[13]=(_Float16)b3.y; b[14]=(_Float16)b3.z; b[15]=(_Float16)b3.w;

    // (neg_a, A, neg_b, B, c_mod, C, reuse_a, reuse_b)
    c = __builtin_amdgcn_wmma_f32_16x16x32_f16(false, a, false, b,
                                               (short)0, c, false, false);
  }

  // C/D layout: VGPR r -> row m0 + r (lanes 0-15) or m0 + 8 + r (lanes 16-31)
  const int crow0 = m0 + grp * 8;
  const int ccol  = n0 + lm;
#pragma unroll
  for (int r = 0; r < 8; ++r) {
    const int row = crow0 + r;
    if (row < N) h[(size_t)row * HC + ccol] = c[r];
  }
}

// ------------------------------------------- per-node attention coeffs -----
// One wave per node: lane handles 8 consecutive channels (lane>>3 = head).
__global__ void gat_node_attn(const float* __restrict__ h,
                              const float* __restrict__ att_s,
                              const float* __restrict__ att_d,
                              float* __restrict__ a_s,
                              float* __restrict__ a_d, int N) {
  const int gid  = blockIdx.x * blockDim.x + threadIdx.x;
  const int n    = gid >> 5;
  const int lane = threadIdx.x & 31;
  if (n >= N) return;
  const int ch = lane * 8;
  const float4 h0 = *(const float4*)(h + (size_t)n * HC + ch);
  const float4 h1 = *(const float4*)(h + (size_t)n * HC + ch + 4);
  const float4 s0 = *(const float4*)(att_s + ch);
  const float4 s1 = *(const float4*)(att_s + ch + 4);
  const float4 d0 = *(const float4*)(att_d + ch);
  const float4 d1 = *(const float4*)(att_d + ch + 4);
  float ss = h0.x*s0.x + h0.y*s0.y + h0.z*s0.z + h0.w*s0.w
           + h1.x*s1.x + h1.y*s1.y + h1.z*s1.z + h1.w*s1.w;
  float dd = h0.x*d0.x + h0.y*d0.y + h0.z*d0.z + h0.w*d0.w
           + h1.x*d1.x + h1.y*d1.y + h1.z*d1.z + h1.w*d1.w;
  // reduce within each 8-lane group (one head per group)
  ss += __shfl_xor(ss, 1, 32); ss += __shfl_xor(ss, 2, 32); ss += __shfl_xor(ss, 4, 32);
  dd += __shfl_xor(dd, 1, 32); dd += __shfl_xor(dd, 2, 32); dd += __shfl_xor(dd, 4, 32);
  if ((lane & 7) == 0) {
    a_s[n * HEADS + (lane >> 3)] = ss;
    a_d[n * HEADS + (lane >> 3)] = dd;
  }
}

__device__ __forceinline__ float lrelu(float v) {
  return v > 0.0f ? v : NEG_SLOPE * v;
}

// --------------------------------------- edge pass 1: exp + denom ----------
// Thread per edge (incl. implicit self-loops e >= E: src = dst = e - E).
__global__ void gat_edge_expsum(const long long* __restrict__ ei,
                                const float* __restrict__ a_s,
                                const float* __restrict__ a_d,
                                float* __restrict__ p,
                                float* __restrict__ denom, int E, int N) {
  const int e = blockIdx.x * blockDim.x + threadIdx.x;
  if (e >= E + N) return;
  int s, d;
  if (e < E) { s = (int)ei[e]; d = (int)ei[E + e]; }
  else       { s = d = e - E; }
  const float4 as = *(const float4*)(a_s + (size_t)s * HEADS);
  const float4 ad = *(const float4*)(a_d + (size_t)d * HEADS);
  float4 pe;
  pe.x = __expf(fminf(lrelu(as.x + ad.x), 30.0f));
  pe.y = __expf(fminf(lrelu(as.y + ad.y), 30.0f));
  pe.z = __expf(fminf(lrelu(as.z + ad.z), 30.0f));
  pe.w = __expf(fminf(lrelu(as.w + ad.w), 30.0f));
  *(float4*)(p + (size_t)e * HEADS) = pe;
  atomicAdd(&denom[(size_t)d * HEADS + 0], pe.x);
  atomicAdd(&denom[(size_t)d * HEADS + 1], pe.y);
  atomicAdd(&denom[(size_t)d * HEADS + 2], pe.z);
  atomicAdd(&denom[(size_t)d * HEADS + 3], pe.w);
}

// --------------------------------------- edge pass 2: weighted scatter -----
// Wave per edge; lane handles 8 consecutive channels (head = lane>>3).
__global__ void gat_edge_message(const long long* __restrict__ ei,
                                 const float* __restrict__ h,
                                 const float* __restrict__ p,
                                 const float* __restrict__ denom,
                                 float* __restrict__ out, int E, int N) {
  const int gid  = blockIdx.x * blockDim.x + threadIdx.x;
  const int e    = gid >> 5;
  const int lane = threadIdx.x & 31;
  if (e >= E + N) return;
  int s, d;
  if (e < E) { s = (int)ei[e]; d = (int)ei[E + e]; }
  else       { s = d = e - E; }
  const int hd = lane >> 3;
  const float alpha = p[(size_t)e * HEADS + hd]
                    / (denom[(size_t)d * HEADS + hd] + 1e-16f);
  const int ch = lane * 8;
  const float4 h0 = *(const float4*)(h + (size_t)s * HC + ch);
  const float4 h1 = *(const float4*)(h + (size_t)s * HC + ch + 4);
  float* o = out + (size_t)d * HC + ch;
  atomicAdd(o + 0, alpha * h0.x);
  atomicAdd(o + 1, alpha * h0.y);
  atomicAdd(o + 2, alpha * h0.z);
  atomicAdd(o + 3, alpha * h0.w);
  atomicAdd(o + 4, alpha * h1.x);
  atomicAdd(o + 5, alpha * h1.y);
  atomicAdd(o + 6, alpha * h1.z);
  atomicAdd(o + 7, alpha * h1.w);
}

// ------------------------------------------------ epilogue -----------------
__global__ void gat_finalize(float* __restrict__ out,
                             const float* __restrict__ x,
                             const float* __restrict__ bias, int total) {
  const int i = blockIdx.x * blockDim.x + threadIdx.x;
  if (i >= total) return;
  float v = out[i] + bias[i & (HC - 1)];
  v = v > 0.0f ? v : expm1f(v);        // ELU(alpha=1)
  out[i] = v + x[i];
}

// ---------------------------------------------------------------------------
extern "C" void kernel_launch(void* const* d_in, const int* in_sizes, int n_in,
                              void* d_out, int out_size, void* d_ws, size_t ws_size,
                              hipStream_t stream) {
  const float*     x     = (const float*)d_in[0];
  const long long* ei    = (const long long*)d_in[1];   // int64 [2, E]
  const float*     W     = (const float*)d_in[2];
  const float*     att_s = (const float*)d_in[3];
  const float*     att_d = (const float*)d_in[4];
  const float*     bias  = (const float*)d_in[5];
  float*           out   = (float*)d_out;

  const int N  = in_sizes[0] / F_IN;   // 50000
  const int E  = in_sizes[1] / 2;      // 800000
  const int E2 = E + N;                // with self loops

  // workspace layout (all 16B-aligned: N*HC, N*HEADS, E2*HEADS all mult of 16)
  float* ws    = (float*)d_ws;
  float* h     = ws;                               // N*HC
  float* a_s   = h     + (size_t)N * HC;           // N*HEADS
  float* a_d   = a_s   + (size_t)N * HEADS;        // N*HEADS
  float* denom = a_d   + (size_t)N * HEADS;        // N*HEADS
  float* p     = denom + (size_t)N * HEADS;        // E2*HEADS

  const int total = N * HC;

  // 1) zero accumulators (d_out is poisoned by the harness)
  gat_init<<<(total + 255) / 256, 256, 0, stream>>>(out, denom, total, N * HEADS);

  // 2) h = x @ W^T via WMMA: one wave per 16x16 tile
  {
    const int mtiles = (N + 15) >> 4;
    const int waves  = mtiles * (HC / 16);
    const int blocks = (waves + 7) / 8;            // 8 waves / 256-thread block
    gat_gemm_wmma<<<blocks, 256, 0, stream>>>(x, W, h, N);
  }

  // 3) per-node attention logits
  gat_node_attn<<<(N + 7) / 8, 256, 0, stream>>>(h, att_s, att_d, a_s, a_d, N);

  // 4) edge softmax numerators + denominators
  gat_edge_expsum<<<(E2 + 255) / 256, 256, 0, stream>>>(ei, a_s, a_d, p, denom, E, N);

  // 5) weighted message scatter-add
  gat_edge_message<<<(E2 + 7) / 8, 256, 0, stream>>>(ei, h, p, denom, out, E, N);

  // 6) bias + ELU + residual
  gat_finalize<<<(total + 255) / 256, 256, 0, stream>>>(out, x, bias, total);
}
